// SkipECell_61890478735978
// MI455X (gfx1250) — matrix operand order repaired and checked by reading, not compiled, not run on hardware
//
#include <hip/hip_runtime.h>
#include <hip/hip_bf16.h>

// ---------------------------------------------------------------------------
// SkipECell (skip-LSTM), T=512, B=128, IN=512, H=1024, SKIP=4  (all fp32 I/O)
// Strategy: bf16 WMMA (v_wmma_f32_16x16x32_bf16) for all matmuls, fp32 accum.
// One fused kernel per timestep; weights resident in L2 (37.7MB << 192MB).
// ---------------------------------------------------------------------------

typedef __attribute__((ext_vector_type(16))) __bf16 v16bf;
typedef __attribute__((ext_vector_type(8)))  __bf16 v8bf;
typedef __attribute__((ext_vector_type(8)))  float  v8f;

#define T_STEPS 512
#define BATCH   128
#define IN_DIM  512
#define H_DIM   1024
#define G_DIM   (4 * H_DIM)            // 4096 gate columns
#define KT      (IN_DIM + 4 * H_DIM)   // 4608 fused K (x | h | hs0 | hs1 | hs2)
#define BH      (BATCH * H_DIM)

__device__ __forceinline__ float sigmoidf_(float x) {
    return 1.0f / (1.0f + __expf(-x));
}

// --- prep: fused bf16 weight matrix Wcat[G_DIM][KT] + summed bias ----------
__global__ void k_prep_wcat(const float* __restrict__ wih,   // [4H, IN]
                            const float* __restrict__ whh,   // [4H, H]
                            const float* __restrict__ wsk,   // [3, 4H, H]
                            const float* __restrict__ bih,
                            const float* __restrict__ bhh,
                            __bf16* __restrict__ Wcat,       // [4H, KT]
                            float* __restrict__ biasS) {
    const int n = blockIdx.x;                 // gate row 0..4095
    __bf16* dst = Wcat + (size_t)n * KT;
    for (int k = threadIdx.x; k < KT; k += blockDim.x) {
        float v;
        if (k < IN_DIM) {
            v = wih[(size_t)n * IN_DIM + k];
        } else if (k < IN_DIM + H_DIM) {
            v = whh[(size_t)n * H_DIM + (k - IN_DIM)];
        } else {
            int kk = k - (IN_DIM + H_DIM);
            int s  = kk / H_DIM;              // 0..2
            int kh = kk - s * H_DIM;
            v = wsk[((size_t)s * G_DIM + n) * H_DIM + kh];
        }
        dst[k] = (__bf16)v;
    }
    if (threadIdx.x == 0) biasS[n] = bih[n] + bhh[n];
}

__global__ void k_f2bf(const float* __restrict__ s, __bf16* __restrict__ d, int n) {
    for (int i = blockIdx.x * blockDim.x + threadIdx.x; i < n;
         i += gridDim.x * blockDim.x)
        d[i] = (__bf16)s[i];
}

__global__ void k_copyf(const float* __restrict__ s, float* __restrict__ d, int n) {
    for (int i = blockIdx.x * blockDim.x + threadIdx.x; i < n;
         i += gridDim.x * blockDim.x)
        d[i] = s[i];
}

// --- fused LSTM step --------------------------------------------------------
// grid: 64 blocks (gate-local column tile c0 = 16*bx, covering cols 0..1023 of
// each of the 4 gates). block: 256 threads = 8 waves; wave w owns batch rows
// [16w, 16w+16). Each wave holds 4 fp32 accumulators (ig/fg/gg/og 16x16 tiles)
// and runs the elementwise cell epilogue in-register.
__global__ __launch_bounds__(256)
void k_step(const __bf16* __restrict__ xt,    // [B, IN]   bf16
            const __bf16* __restrict__ h1,    // [B, H]    h_{t-1}
            const __bf16* __restrict__ h4,    // h_{t-4} (valid iff skipActive)
            const __bf16* __restrict__ h3,    // h_{t-3}
            const __bf16* __restrict__ h2,    // h_{t-2}
            int skipActive,
            const __bf16* __restrict__ Wcat,  // [G, KT]
            const float*  __restrict__ bias,  // [G]
            float*  __restrict__ c_state,     // [B, H] fp32 (read+write)
            __bf16* __restrict__ histW,       // [B, H] bf16 ring slot (write)
            float*  __restrict__ outT,        // [B, H] fp32 output slice
            float*  __restrict__ hFinal,      // nullptr except last step
            float*  __restrict__ cFinal) {
    const int tid    = threadIdx.x;
    const int lane   = tid & 31;
    const int wv     = tid >> 5;          // 0..7 -> M tile
    const int m0     = wv * 16;
    const int c0     = blockIdx.x * 16;   // gate-local column tile base
    const int ln     = lane & 15;
    const bool hiH   = lane >= 16;

    v8f acc0 = {}, acc1 = {}, acc2 = {}, acc3 = {};

    // B-fragment row pointers for the 4 gates (cols c0+j*1024+ln), contiguous-k layout (this code's order, not the ISA table order):
    // lanes 0-15 carry K=0..15 of the K-block, lanes 16-31 carry K=16..31,
    // each as 16 contiguous bf16 (32B) from the row-major weight row.
    const __bf16* bp0 = Wcat + (size_t)(c0 + ln) * KT + (hiH ? 16 : 0);
    const __bf16* bp1 = bp0 + (size_t)H_DIM * KT;
    const __bf16* bp2 = bp1 + (size_t)H_DIM * KT;
    const __bf16* bp3 = bp2 + (size_t)H_DIM * KT;

    // K segments over the fused activation row [x | h1 | h4 | h3 | h2]
    const __bf16* segSrc[5] = { xt, h1, h4, h3, h2 };
    const int segLd[5]  = { IN_DIM, H_DIM, H_DIM, H_DIM, H_DIM };
    const int segW0[5]  = { 0, IN_DIM, IN_DIM + H_DIM, IN_DIM + 2 * H_DIM,
                            IN_DIM + 3 * H_DIM };
    const int nSkipKb   = skipActive ? (H_DIM / 32) : 0;
    const int segNkb[5] = { IN_DIM / 32, H_DIM / 32, nSkipKb, nSkipKb, nSkipKb };

    for (int s = 0; s < 5; ++s) {
        const int nkb = segNkb[s];
        if (nkb == 0) continue;
        // A-fragment (matches B): lanes 0-15 take K 0..15,
        // lanes 16-31 take K 16..31; two 16B contiguous loads.
        const __bf16* arow = segSrc[s] + (size_t)(m0 + ln) * segLd[s] + (hiH ? 16 : 0);
        int wk = segW0[s];
        for (int kb = 0; kb < nkb; ++kb, wk += 32) {
            const int k0 = kb * 32;
            v8bf alo = *(const v8bf*)(arow + k0);
            v8bf ahi = *(const v8bf*)(arow + k0 + 8);
            v16bf a;
#pragma unroll
            for (int i = 0; i < 8; ++i) { a[i] = alo[i]; a[8 + i] = ahi[i]; }

            v16bf b0 = *(const v16bf*)(bp0 + wk);
            v16bf b1 = *(const v16bf*)(bp1 + wk);
            v16bf b2 = *(const v16bf*)(bp2 + wk);
            v16bf b3 = *(const v16bf*)(bp3 + wk);

            // keep the L2->WGP weight stream ahead of the WMMA pipe
            __builtin_prefetch(bp0 + wk + 32, 0, 1);
            __builtin_prefetch(bp2 + wk + 32, 0, 1);

            acc0 = __builtin_amdgcn_wmma_f32_16x16x32_bf16(false, a, false, b0,
                                                           (short)0, acc0, false, false);
            acc1 = __builtin_amdgcn_wmma_f32_16x16x32_bf16(false, a, false, b1,
                                                           (short)0, acc1, false, false);
            acc2 = __builtin_amdgcn_wmma_f32_16x16x32_bf16(false, a, false, b2,
                                                           (short)0, acc2, false, false);
            acc3 = __builtin_amdgcn_wmma_f32_16x16x32_bf16(false, a, false, b3,
                                                           (short)0, acc3, false, false);
        }
    }

    // --- fused LSTM cell epilogue (C/D layout: VGPR r -> M = r or 8+r) -----
    const int colg = c0 + ln;                 // hidden column 0..1023
    const float bi = bias[colg];
    const float bf_ = bias[colg + H_DIM];
    const float bg = bias[colg + 2 * H_DIM];
    const float bo = bias[colg + 3 * H_DIM];

#pragma unroll
    for (int r = 0; r < 8; ++r) {
        const int row = m0 + (hiH ? (8 + r) : r);
        const size_t idx = (size_t)row * H_DIM + colg;
        const float ig = acc0[r] + bi;
        const float fg = acc1[r] + bf_;
        const float gg = acc2[r] + bg;
        const float og = acc3[r] + bo;
        const float iv = sigmoidf_(ig);
        const float fv = sigmoidf_(fg);
        const float gv = tanhf(gg);
        const float ov = sigmoidf_(og);
        const float cn = fv * c_state[idx] + iv * gv;
        const float hv = ov * tanhf(cn);
        c_state[idx] = cn;
        outT[idx]    = hv;
        histW[idx]   = (__bf16)hv;
        if (hFinal) { hFinal[idx] = hv; cFinal[idx] = cn; }
    }
}

// ---------------------------------------------------------------------------
extern "C" void kernel_launch(void* const* d_in, const int* in_sizes, int n_in,
                              void* d_out, int out_size, void* d_ws, size_t ws_size,
                              hipStream_t stream) {
    (void)in_sizes; (void)n_in; (void)out_size; (void)ws_size;
    const float* x         = (const float*)d_in[0];  // [T, B, IN]
    const float* hidden    = (const float*)d_in[1];  // [B, H]
    const float* candidate = (const float*)d_in[2];  // [B, H]
    const float* wih       = (const float*)d_in[3];  // [4H, IN]
    const float* whh       = (const float*)d_in[4];  // [4H, H]
    const float* bih       = (const float*)d_in[5];  // [4H]
    const float* bhh       = (const float*)d_in[6];  // [4H]
    const float* wsk       = (const float*)d_in[7];  // [3, 4H, H]

    // workspace carve-up (256B aligned): ~103 MB total
    char* ws = (char*)d_ws;
    size_t off = 0;
    auto carve = [&](size_t bytes) -> void* {
        void* p = ws + off;
        off = (off + bytes + 255) & ~(size_t)255;
        return p;
    };
    __bf16* Wcat = (__bf16*)carve((size_t)G_DIM * KT * 2);              // 37.7MB
    __bf16* xbf  = (__bf16*)carve((size_t)T_STEPS * BATCH * IN_DIM * 2);// 64MB
    __bf16* hist = (__bf16*)carve((size_t)8 * BH * 2);                  // 2MB ring
    float*  biasS = (float*)carve((size_t)G_DIM * 4);
    float*  cst   = (float*)carve((size_t)BH * 4);

    float* out  = (float*)d_out;                 // [T, B, H]
    float* hFin = out + (size_t)T_STEPS * BH;    // [B, H]
    float* cFin = hFin + BH;                     // [B, H]

    // --- one-time prep ---
    k_prep_wcat<<<G_DIM, 256, 0, stream>>>(wih, whh, wsk, bih, bhh, Wcat, biasS);
    {
        const int n = T_STEPS * BATCH * IN_DIM;
        k_f2bf<<<8192, 256, 0, stream>>>(x, xbf, n);
    }
    k_f2bf<<<512, 256, 0, stream>>>(hidden, hist + (size_t)7 * BH, BH); // slot (-1)&7
    k_copyf<<<512, 256, 0, stream>>>(candidate, cst, BH);

    // --- sequential recurrence: one fused kernel per timestep ---
    int qsize = 0;  // reference's queue-size state machine
    for (int t = 0; t < T_STEPS; ++t) {
        const bool full = (qsize == 4);
        const __bf16* h1 = hist + (size_t)((t - 1) & 7) * BH;
        const __bf16* h4 = full ? hist + (size_t)((t - 4) & 7) * BH : h1;
        const __bf16* h3 = full ? hist + (size_t)((t - 3) & 7) * BH : h1;
        const __bf16* h2 = full ? hist + (size_t)((t - 2) & 7) * BH : h1;
        k_step<<<dim3(64), dim3(256), 0, stream>>>(
            xbf + (size_t)t * BATCH * IN_DIM, h1, h4, h3, h2, full ? 1 : 0,
            Wcat, biasS, cst, hist + (size_t)(t & 7) * BH,
            out + (size_t)t * BH,
            (t == T_STEPS - 1) ? hFin : nullptr, cFin);
        qsize = full ? 2 : qsize + 1;
    }
}
